// MultiHeadSelfAttention_72997264162956
// MI455X (gfx1250) — matrix operand order repaired and checked by reading, not compiled, not run on hardware
//
#include <hip/hip_runtime.h>
#include <hip/hip_bf16.h>
#include <math.h>

// ---------------------------------------------------------------------------
// MultiHeadSelfAttention for MI455X (gfx1250, wave32, WMMA 16x16x32 f16,
// async global->LDS staging with ASYNCcnt).
// b=2, n=2048, EMB=1024, HEADS=16, HEAD_DIM=64
// ---------------------------------------------------------------------------

typedef _Float16 h16;
typedef __attribute__((ext_vector_type(16))) _Float16 v16h;
typedef __attribute__((ext_vector_type(8)))  _Float16 v8h;
typedef __attribute__((ext_vector_type(8)))  float    v8f;
typedef __attribute__((ext_vector_type(4)))  float    v4f;

#define EMB   1024
#define SEQ   2048
#define BATCH 2
#define HEADS 16
#define HD    64
#define ROWS  (BATCH * SEQ)          // 4096 token rows

static __device__ __forceinline__ v16h cat8(v8h a, v8h b) {
    return __builtin_shufflevector(a, b, 0,1,2,3,4,5,6,7,8,9,10,11,12,13,14,15);
}

// D = A(16x32 f16) * B(32x16 f16) + C(16x16 f32)
static __device__ __forceinline__ v8f wmma32(v16h a, v16h b, v8f c) {
    return __builtin_amdgcn_wmma_f32_16x16x32_f16(
        /*neg_a=*/false, a, /*neg_b=*/false, b,
        /*c_mod=*/(short)0, c, /*reuse_a=*/false, /*reuse_b=*/false);
}

// Async 16B copy global -> LDS (gfx1250 GLOBAL_LOAD_ASYNC_TO_LDS_B128, ASYNCcnt).
static __device__ __forceinline__ void async_copy16(h16* ldst, const h16* gsrc) {
    unsigned loff = (unsigned)(unsigned long long)(const void*)ldst;  // LDS byte addr
    unsigned long long gaddr = (unsigned long long)gsrc;
    asm volatile("global_load_async_to_lds_b128 %0, %1, off"
                 :: "v"(loff), "v"(gaddr) : "memory");
}
static __device__ __forceinline__ void wait_async0() {
    asm volatile("s_wait_asynccnt 0" ::: "memory");
}

// A-fragment (16x32, f16) from fp32 row-major source. row0 points at (row 0, k0).
// Lane layout: M = lane%16 ; elems 0..7 -> K = 8*(lane/16)+0..7 ,
//              elems 8..15 -> K = 16 + 8*(lane/16) + 0..7   (ISA 7.12.2)
static __device__ __forceinline__ v16h load_a_f32(const float* row0, int ld, int lane) {
    int m = lane & 15, hl = lane >> 4;
    const float* p = row0 + (size_t)m * ld + 8 * hl;
    v4f f0 = *(const v4f*)(p);
    v4f f1 = *(const v4f*)(p + 4);
    v4f f2 = *(const v4f*)(p + 16);
    v4f f3 = *(const v4f*)(p + 20);
    v8f lo = __builtin_shufflevector(f0, f1, 0,1,2,3,4,5,6,7);
    v8f hi = __builtin_shufflevector(f2, f3, 0,1,2,3,4,5,6,7);
    return cat8(__builtin_convertvector(lo, v8h), __builtin_convertvector(hi, v8h));
}

// A-fragment (16x32, f16) from f16 row-major source (global or LDS).
static __device__ __forceinline__ v16h load_a_f16(const h16* row0, int ld, int lane) {
    int m = lane & 15, hl = lane >> 4;
    const h16* p = row0 + (size_t)m * ld + 8 * hl;
    v8h c0 = *(const v8h*)(p);
    v8h c1 = *(const v8h*)(p + 16);
    return cat8(c0, c1);
}

// B-fragment (32x16) from [N][K]-ordered f16 tile (row n holds the K-run of col n).
// Lane layout: N = lane%16 ; elems 0..7 -> K = k0 + 8*(lane/16)+0..7 , elems 8..15 -> K = k0 + 16 + 8*(lane/16) + 0..7   (ISA 7.12.2)
static __device__ __forceinline__ v16h load_b_nk(const h16* t, int ld, int n0, int k0, int lane) {
    int n = lane & 15, hl = lane >> 4;
    const h16* p = t + (size_t)(n0 + n) * ld + k0 + 8 * hl;
    v8h c0 = *(const v8h*)(p);
    v8h c1 = *(const v8h*)(p + 16);
    return cat8(c0, c1);
}

// ---------------------------------------------------------------------------
// Kernel 1: QKV projection.  out = x @ W^T + b, written f16 in [b,h,n,d] order.
// grid = (ROWS/256, EMB/64, 3), block = 256 (8 waves; wave w owns rows 32w..32w+31)
// ---------------------------------------------------------------------------
__global__ __launch_bounds__(256)
void qkv_proj_kernel(const float* __restrict__ x,
                     const float* __restrict__ Wq, const float* __restrict__ bq,
                     const float* __restrict__ Wk, const float* __restrict__ bk,
                     const float* __restrict__ Wv, const float* __restrict__ bv,
                     h16* __restrict__ qh, h16* __restrict__ kh, h16* __restrict__ vh) {
    __shared__ h16 ldsB[64 * 32];             // W tile in [N][K] order

    const int tid  = threadIdx.x;
    const int lane = tid & 31;
    const int wv   = tid >> 5;
    const int z    = blockIdx.z;
    const float* W    = (z == 0) ? Wq : (z == 1) ? Wk : Wv;
    const float* bias = (z == 0) ? bq : (z == 1) ? bk : bv;
    h16*         dst  = (z == 0) ? qh : (z == 1) ? kh : vh;

    const int rowBase = blockIdx.x * 256 + wv * 32;
    const int nBase   = blockIdx.y * 64;     // == head * 64

    v8f acc[2][4] = {};

    for (int k = 0; k < EMB; k += 32) {
        // stage 64x32 W tile (rows of W are contiguous in K -> straight copy+cvt)
        for (int i = tid; i < 64 * 32; i += 256) {
            int r = i >> 5, c = i & 31;
            ldsB[i] = (h16)W[(size_t)(nBase + r) * EMB + k + c];
        }
        __syncthreads();

        v16h a0 = load_a_f32(x + (size_t)rowBase * EMB + k, EMB, lane);
        v16h a1 = load_a_f32(x + (size_t)(rowBase + 16) * EMB + k, EMB, lane);
#pragma unroll
        for (int f = 0; f < 4; ++f) {
            v16h b = load_b_nk(ldsB, 32, 16 * f, 0, lane);
            acc[0][f] = wmma32(a0, b, acc[0][f]);
            acc[1][f] = wmma32(a1, b, acc[1][f]);
        }
        __syncthreads();
    }

    const int hl = lane >> 4, nn = lane & 15;
    const int h  = blockIdx.y;
#pragma unroll
    for (int mi = 0; mi < 2; ++mi) {
#pragma unroll
        for (int f = 0; f < 4; ++f) {
#pragma unroll
            for (int r = 0; r < 8; ++r) {
                int M    = mi * 16 + r + 8 * hl;
                int grow = rowBase + M;
                int batch = grow >> 11, pos = grow & (SEQ - 1);
                int d    = 16 * f + nn;
                float v  = acc[mi][f][r] + bias[nBase + d];
                dst[(((size_t)batch * HEADS + h) * SEQ + pos) * HD + d] = (h16)v;
            }
        }
    }
}

// ---------------------------------------------------------------------------
// Kernel 2: flash attention per (b,h).  grid = (SEQ/128, BATCH*HEADS), block=256.
// Wave w owns 16 query rows; K/V streamed in double-buffered 32-key LDS tiles,
// K tile staged with async global->LDS, V tile staged transposed by threads.
// ---------------------------------------------------------------------------
__global__ __launch_bounds__(256)
void attn_kernel(const h16* __restrict__ qh, const h16* __restrict__ kh,
                 const h16* __restrict__ vh, h16* __restrict__ ctx) {
    __shared__ h16 Kt[2][32 * 64];            // [key][d]   (= [N][K] for scores)
    __shared__ h16 Vt[2][64 * 32];            // [d][key]   (= [N][K] for ctx)
    __shared__ h16 Pt[8][16 * 32];            // per-wave P transpose buffer

    const int tid  = threadIdx.x;
    const int lane = tid & 31;
    const int wv   = tid >> 5;
    const int hl   = lane >> 4, nn = lane & 15;
    const int bh   = blockIdx.y;
    const int batch = bh >> 4, h = bh & 15;
    const float scale = 0.125f;               // 1/sqrt(64)

    const size_t headOff = (size_t)bh * SEQ * HD;
    const int q0 = blockIdx.x * 128 + 16 * wv;

    // Q fragments stay in registers for the whole pass (d = 0..31, 32..63)
    const h16* qrow = qh + headOff + (size_t)q0 * HD;
    const v16h aq0 = load_a_f16(qrow,      HD, lane);
    const v16h aq1 = load_a_f16(qrow + 32, HD, lane);

    float mrow[8], lrow[8];
    v8f acc[4] = {};
#pragma unroll
    for (int r = 0; r < 8; ++r) { mrow[r] = -1e30f; lrow[r] = 0.f; }

    // ---- prologue: stage tile 0 into buffer 0 ----
    async_copy16(&Kt[0][tid * 8], kh + headOff + tid * 8);     // 4KB async copy
    {
        const h16* src = vh + headOff;
        for (int i = tid; i < 32 * 64; i += 256) {
            int m = i >> 6, d = i & 63;
            Vt[0][d * 32 + m] = src[i];
        }
    }
    wait_async0();
    __syncthreads();

    for (int kt = 0; kt < SEQ; kt += 32) {
        const int buf = (kt >> 5) & 1, nxt = buf ^ 1;

        // ---- stage next tile while computing on the current one ----
        if (kt + 32 < SEQ) {
            async_copy16(&Kt[nxt][tid * 8],
                         kh + headOff + (size_t)(kt + 32) * HD + tid * 8);
            const h16* src = vh + headOff + (size_t)(kt + 32) * HD;
            for (int i = tid; i < 32 * 64; i += 256) {
                int m = i >> 6, d = i & 63;
                Vt[nxt][d * 32 + m] = src[i];
            }
        }
        if (kt + 64 < SEQ) {  // gfx1250 global_prefetch_b8 two tiles ahead
            __builtin_prefetch(kh + headOff + (size_t)(kt + 64) * HD + tid * 8, 0, 0);
            __builtin_prefetch(vh + headOff + (size_t)(kt + 64) * HD + tid * 8, 0, 0);
        }

        // ---- scores: S(16x32) = Q(16x64) . K^T ----
        v8f c0 = {}, c1 = {};
        c0 = wmma32(aq0, load_b_nk(Kt[buf], 64,  0,  0, lane), c0);
        c0 = wmma32(aq1, load_b_nk(Kt[buf], 64,  0, 32, lane), c0);
        c1 = wmma32(aq0, load_b_nk(Kt[buf], 64, 16,  0, lane), c1);
        c1 = wmma32(aq1, load_b_nk(Kt[buf], 64, 16, 32, lane), c1);

        // ---- online softmax (row stats across the 16-lane column groups) ----
        float alpha[8];
#pragma unroll
        for (int r = 0; r < 8; ++r) {
            float s0 = c0[r] * scale, s1 = c1[r] * scale;
            float mx = fmaxf(s0, s1);
#pragma unroll
            for (int msk = 1; msk < 16; msk <<= 1)
                mx = fmaxf(mx, __shfl_xor(mx, msk, 32));
            float mnew = fmaxf(mrow[r], mx);
            float p0 = __expf(s0 - mnew);
            float p1 = __expf(s1 - mnew);
            float ps = p0 + p1;
#pragma unroll
            for (int msk = 1; msk < 16; msk <<= 1)
                ps += __shfl_xor(ps, msk, 32);
            alpha[r] = __expf(mrow[r] - mnew);
            lrow[r]  = lrow[r] * alpha[r] + ps;
            mrow[r]  = mnew;
            int row = r + 8 * hl;                       // C-layout row of this elem
            Pt[wv][row * 32 + nn]      = (h16)p0;
            Pt[wv][row * 32 + 16 + nn] = (h16)p1;
        }
#pragma unroll
        for (int f = 0; f < 4; ++f)
#pragma unroll
            for (int r = 0; r < 8; ++r) acc[f][r] *= alpha[r];

        // ---- ctx += P(16x32) . V(32x64) ----
        v16h ap = load_a_f16(&Pt[wv][0], 32, lane);
#pragma unroll
        for (int f = 0; f < 4; ++f)
            acc[f] = wmma32(ap, load_b_nk(Vt[buf], 32, 16 * f, 0, lane), acc[f]);

        wait_async0();        // next K tile resident in LDS
        __syncthreads();      // all waves done with current buffers
    }

    // ---- normalize and write ctx in [b, n, h*64+d] (merged-head) layout ----
#pragma unroll
    for (int f = 0; f < 4; ++f) {
#pragma unroll
        for (int r = 0; r < 8; ++r) {
            int M   = r + 8 * hl;
            int pos = q0 + M;
            int d   = 16 * f + nn;
            float v = acc[f][r] / lrow[r];
            ctx[((size_t)batch * SEQ + pos) * EMB + h * HD + d] = (h16)v;
        }
    }
}

// ---------------------------------------------------------------------------
// Kernel 3: output projection.  out = ctx @ Wo^T + bo  (fp32 out)
// grid = (ROWS/256, EMB/64), block = 256 (wave owns 32 rows)
// ---------------------------------------------------------------------------
__global__ __launch_bounds__(256)
void out_proj_kernel(const h16* __restrict__ ctx, const float* __restrict__ Wo,
                     const float* __restrict__ bo, float* __restrict__ out) {
    __shared__ h16 ldsB[64 * 32];

    const int tid  = threadIdx.x;
    const int lane = tid & 31;
    const int wv   = tid >> 5;
    const int rowBase = blockIdx.x * 256 + wv * 32;
    const int nBase   = blockIdx.y * 64;

    v8f acc[2][4] = {};

    for (int k = 0; k < EMB; k += 32) {
        for (int i = tid; i < 64 * 32; i += 256) {
            int r = i >> 5, c = i & 31;
            ldsB[i] = (h16)Wo[(size_t)(nBase + r) * EMB + k + c];
        }
        __syncthreads();

        v16h a0 = load_a_f16(ctx + (size_t)rowBase * EMB + k, EMB, lane);
        v16h a1 = load_a_f16(ctx + (size_t)(rowBase + 16) * EMB + k, EMB, lane);
#pragma unroll
        for (int f = 0; f < 4; ++f) {
            v16h b = load_b_nk(ldsB, 32, 16 * f, 0, lane);
            acc[0][f] = wmma32(a0, b, acc[0][f]);
            acc[1][f] = wmma32(a1, b, acc[1][f]);
        }
        __syncthreads();
    }

    const int hl = lane >> 4, nn = lane & 15;
#pragma unroll
    for (int mi = 0; mi < 2; ++mi) {
#pragma unroll
        for (int f = 0; f < 4; ++f) {
#pragma unroll
            for (int r = 0; r < 8; ++r) {
                int M = mi * 16 + r + 8 * hl;
                int d = 16 * f + nn;
                out[(size_t)(rowBase + M) * EMB + nBase + d] =
                    acc[mi][f][r] + bo[nBase + d];
            }
        }
    }
}

// ---------------------------------------------------------------------------
extern "C" void kernel_launch(void* const* d_in, const int* in_sizes, int n_in,
                              void* d_out, int out_size, void* d_ws, size_t ws_size,
                              hipStream_t stream) {
    const float* x  = (const float*)d_in[0];
    const float* Wq = (const float*)d_in[1];
    const float* bq = (const float*)d_in[2];
    const float* Wk = (const float*)d_in[3];
    const float* bk = (const float*)d_in[4];
    const float* Wv = (const float*)d_in[5];
    const float* bv = (const float*)d_in[6];
    const float* Wo = (const float*)d_in[7];
    const float* bo = (const float*)d_in[8];
    float* out = (float*)d_out;

    const size_t perTensor = (size_t)BATCH * HEADS * SEQ * HD;  // 4,194,304 halfs
    h16* ws  = (h16*)d_ws;
    h16* qh  = ws;
    h16* kh  = ws + perTensor;
    h16* vh  = ws + 2 * perTensor;
    h16* ctx = ws + 3 * perTensor;   // total 32 MB of f16 scratch

    qkv_proj_kernel<<<dim3(ROWS / 256, EMB / 64, 3), 256, 0, stream>>>(
        x, Wq, bq, Wk, bk, Wv, bv, qh, kh, vh);
    attn_kernel<<<dim3(SEQ / 128, BATCH * HEADS), 256, 0, stream>>>(qh, kh, vh, ctx);
    out_proj_kernel<<<dim3(ROWS / 256, EMB / 64), 256, 0, stream>>>(ctx, Wo, bo, out);

    (void)in_sizes; (void)n_in; (void)out_size; (void)ws_size;
}
